// HeteroSAGE_18769007083672
// MI455X (gfx1250) — matrix operand order repaired and checked by reading, not compiled, not run on hardware
//
#include <hip/hip_runtime.h>
#include <hip/hip_bf16.h>

// ---------------------------------------------------------------------------
// HeteroSAGE on MI455X (gfx1250): bf16 activations, v_wmma_f32_16x16x32_bf16
// GEMM core with fragment-major LDS weight layout (B frag = 2x ds_load_b128),
// f32 atomic scatter-mean for card/merch aggregation, coalesced b128 output.
// ---------------------------------------------------------------------------

#define H      128
#define H2     64          // packed bf16-pairs per full-K row
#define NTX    200000
#define NCARD  50000
#define NMERCH 10000
#define NEDGE  200000

typedef __attribute__((ext_vector_type(16))) __bf16 v16bf;
typedef __attribute__((ext_vector_type(8)))  float  v8f;

union FragBF { v16bf v; unsigned u[8]; uint4 q[2]; };

__device__ __forceinline__ unsigned short f32_bf16(float f) {
  unsigned u = __float_as_uint(f);
  u += 0x7FFFu + ((u >> 16) & 1u);           // round-to-nearest-even
  return (unsigned short)(u >> 16);
}
__device__ __forceinline__ unsigned pk_bf16(float lo, float hi) {
  return (unsigned)f32_bf16(lo) | ((unsigned)f32_bf16(hi) << 16);
}
__device__ __forceinline__ float bf16_f32(unsigned short h) {
  return __uint_as_float(((unsigned)h) << 16);
}

// ---------------------------------------------------------------------------
// Wave-level WMMA core: acc[8] (16x128 f32 strip) += A(16xK) * W(KxH).
//
// al: LDS packed A rows, al[row*K2 + k2] = {bf16 A[row][2k2], A[row][2k2+1]}.
//     A-fragment (16-bit A 16x32, contiguous k = 16*half + s, same as B): lane holds row = lane&15,
//     K-pairs k2 = c*16 + (lane>>4)*8 + v  -> 8 contiguous dwords
//     at [c*16 + half*8]: two b128 LDS loads.
//
// wl: LDS weights pre-packed *fragment-major* by pack_w_kernel:
//     wl[((c*8 + t)*32 + lane)*8 + v] = {bf16 W[2k2][n], W[2k2+1][n]}
//     with k2 = c*16 + (lane>>4)*8 + v, n = t*16 + (lane&15)
//     -> each lane's B fragment is 8 contiguous dwords: two b128 LDS loads.
// ---------------------------------------------------------------------------
__device__ __forceinline__ void wmma_accum(const unsigned* wl, const unsigned* al,
                                           v8f acc[8], int kChunks, int K2) {
  const int lane = threadIdx.x & 31;
  const int half = lane >> 4;
  const int row  = lane & 15;
  const uint4* w4 = (const uint4*)wl;
  for (int c = 0; c < kChunks; ++c) {
    FragBF A;
    const uint4* a4 = (const uint4*)(al + row * K2 + c * 16 + half * 8);
    A.q[0] = a4[0];
    A.q[1] = a4[1];          // +4 dwords
#pragma unroll
    for (int t = 0; t < 8; ++t) {
      FragBF B;
      int bi = ((c * 8 + t) * 32 + lane) * 2;   // uint4 index
      B.q[0] = w4[bi];
      B.q[1] = w4[bi + 1];
      acc[t] = __builtin_amdgcn_wmma_f32_16x16x32_bf16(
          false, A.v, false, B.v, (short)0, acc[t], false, false);
    }
  }
}

// Cooperative (whole block) LDS fill of packed weights, 16B vectors.
__device__ __forceinline__ void load_w(unsigned* dst, const unsigned* src, int n4) {
  uint4* d4 = (uint4*)dst;
  const uint4* s4 = (const uint4*)src;
  for (int i = threadIdx.x; i < n4; i += blockDim.x) d4[i] = s4[i];
}

// Stage 16 rows of a bf16 matrix (pair-packed uints) into this wave's LDS
// region with 16B vectors; optional row gather through gidx.
__device__ __forceinline__ void stage_bf16(unsigned* al, const unsigned* src,
                                           const int* gidx, int rowBase, int M, int K2) {
  const int lane = threadIdx.x & 31;
  const int Q = K2 >> 2;                    // uint4 per row
  for (int idx = lane; idx < 16 * Q; idx += 32) {
    int r = idx / Q, q = idx - r * Q;
    int grow = rowBase + r;
    uint4 val = {0u, 0u, 0u, 0u};
    if (grow < M) {
      int srow = gidx ? gidx[grow] : grow;
      val = ((const uint4*)src)[(size_t)srow * Q + q];
    }
    ((uint4*)al)[r * Q + q] = val;
  }
}

// Stage 16 rows of an f32 matrix with optional per-row 1/max(cnt,1) scaling,
// converting to packed bf16 pairs (8 floats -> one uint4 per step).
__device__ __forceinline__ void stage_f32(unsigned* al, const float* src,
                                          const float* cnt, int rowBase, int M, int K2) {
  const int lane = threadIdx.x & 31;
  const int Q = K2 >> 2;
  for (int idx = lane; idx < 16 * Q; idx += 32) {
    int r = idx / Q, q = idx - r * Q;
    int grow = rowBase + r;
    uint4 val = {0u, 0u, 0u, 0u};
    if (grow < M) {
      float s = cnt ? 1.0f / fmaxf(cnt[grow], 1.0f) : 1.0f;
      const float4* p = (const float4*)(src + (size_t)grow * (K2 * 2)) + q * 2;
      float4 f0 = p[0], f1 = p[1];
      val.x = pk_bf16(f0.x * s, f0.y * s);
      val.y = pk_bf16(f0.z * s, f0.w * s);
      val.z = pk_bf16(f1.x * s, f1.y * s);
      val.w = pk_bf16(f1.z * s, f1.w * s);
    }
    ((uint4*)al)[r * Q + q] = val;
  }
}

// Bias + (relu) + bf16 convert, staged through LDS (reusing the A-staging
// region: 16 rows x 128 bf16 = 1024 dwords) then coalesced b128 stores.
__device__ __forceinline__ void store_tile(unsigned short* out, const v8f acc[8],
                                           const float* bias, int rowBase, int M,
                                           bool relu, unsigned* stage) {
  const int lane = threadIdx.x & 31;
  const int half = lane >> 4, col = lane & 15;
  unsigned short* s = (unsigned short*)stage;
#pragma unroll
  for (int t = 0; t < 8; ++t) {
    int n = t * 16 + col;
    float b = bias ? bias[n] : 0.f;
#pragma unroll
    for (int r = 0; r < 8; ++r) {
      float v = acc[t][r] + b;
      if (relu) v = fmaxf(v, 0.f);
      s[(half * 8 + r) * H + n] = f32_bf16(v);
    }
  }
  __syncthreads();
  const uint4* s4 = (const uint4*)stage;
  for (int idx = lane; idx < 256; idx += 32) {   // 16 rows x 16 uint4
    int r = idx >> 4, q = idx & 15;
    int m = rowBase + r;
    if (m < M) ((uint4*)(out + (size_t)m * H))[q] = s4[idx];
  }
}

// ---------------------------------------------------------------------------
// Kernels
// ---------------------------------------------------------------------------

// out = act(X_f32 @ W + b)      (input encoders; K = 64 or 128)
__global__ __launch_bounds__(128) void encoder_kernel(
    const float* __restrict__ X, const unsigned* __restrict__ W,
    const float* __restrict__ bias, unsigned short* __restrict__ out,
    int M, int K2, int doRelu) {
  __shared__ __align__(16) unsigned wl[H2 * H];
  __shared__ __align__(16) unsigned al[4][16 * H2];
  const int wave = threadIdx.x >> 5;
  const int rowBase = blockIdx.x * 64 + wave * 16;
  load_w(wl, W, (K2 * H) / 4);
  stage_f32(al[wave], X, nullptr, rowBase, M, K2);
  __syncthreads();
  v8f acc[8];
#pragma unroll
  for (int t = 0; t < 8; ++t) acc[t] = (v8f)(0.0f);
  wmma_accum(wl, al[wave], acc, K2 / 16, K2);
  store_tile(out, acc, bias, rowBase, M, doRelu != 0, al[wave]);
}

// h_tx' = relu(hC[eC] @ Wl0 + hM[eM] @ Wl2 + hT @ (Wr0+Wr2) + (bl0+bl2))
__global__ __launch_bounds__(128) void tx_layer_kernel(
    const unsigned* __restrict__ hC, const unsigned* __restrict__ hM,
    const unsigned* __restrict__ hT,
    const int* __restrict__ eC, const int* __restrict__ eM,
    const unsigned* __restrict__ W0, const unsigned* __restrict__ W2,
    const unsigned* __restrict__ Wrc, const float* __restrict__ bias,
    unsigned short* __restrict__ out, int M) {
  __shared__ __align__(16) unsigned wl[H2 * H];
  __shared__ __align__(16) unsigned al[4][16 * H2];
  const int wave = threadIdx.x >> 5;
  const int rowBase = blockIdx.x * 64 + wave * 16;
  const unsigned* srcs[3] = {hC, hM, hT};
  const int*      gs[3]   = {eC, eM, nullptr};
  const unsigned* Ws[3]   = {W0, W2, Wrc};
  v8f acc[8];
#pragma unroll
  for (int t = 0; t < 8; ++t) acc[t] = (v8f)(0.0f);
#pragma unroll
  for (int op = 0; op < 3; ++op) {
    __syncthreads();   // previous op's LDS reads complete before overwrite
    load_w(wl, Ws[op], (H2 * H) / 4);
    stage_bf16(al[wave], srcs[op], gs[op], rowBase, M, H2);
    __syncthreads();
    wmma_accum(wl, al[wave], acc, 4, H2);
  }
  store_tile(out, acc, bias, rowBase, M, true, al[wave]);
}

// h_dst' = relu((agg/max(cnt,1)) @ Wl + h_dst @ Wr + bl)
__global__ __launch_bounds__(128) void dst_layer_kernel(
    const float* __restrict__ agg, const float* __restrict__ cnt,
    const unsigned* __restrict__ hD,
    const unsigned* __restrict__ Wl, const unsigned* __restrict__ Wr,
    const float* __restrict__ bias, unsigned short* __restrict__ out, int M) {
  __shared__ __align__(16) unsigned wl[H2 * H];
  __shared__ __align__(16) unsigned al[4][16 * H2];
  const int wave = threadIdx.x >> 5;
  const int rowBase = blockIdx.x * 64 + wave * 16;
  v8f acc[8];
#pragma unroll
  for (int t = 0; t < 8; ++t) acc[t] = (v8f)(0.0f);
  // op 0: mean-aggregated neighbors (f32 + per-row scale)
  load_w(wl, Wl, (H2 * H) / 4);
  stage_f32(al[wave], agg, cnt, rowBase, M, H2);
  __syncthreads();
  wmma_accum(wl, al[wave], acc, 4, H2);
  __syncthreads();
  // op 1: self features
  load_w(wl, Wr, (H2 * H) / 4);
  stage_bf16(al[wave], hD, nullptr, rowBase, M, H2);
  __syncthreads();
  wmma_accum(wl, al[wave], acc, 4, H2);
  store_tile(out, acc, bias, rowBase, M, true, al[wave]);
}

// logits = relu(hT @ h1_W + h1_b) @ h2_W + h2_b   (fused GEMM + dot-reduce)
__global__ __launch_bounds__(128) void head_kernel(
    const unsigned* __restrict__ hT, const unsigned* __restrict__ Wh1,
    const float* __restrict__ b1, const float* __restrict__ w2,
    const float* __restrict__ b2, float* __restrict__ out, int M) {
  __shared__ __align__(16) unsigned wl[H2 * H];
  __shared__ __align__(16) unsigned al[4][16 * H2];
  const int wave = threadIdx.x >> 5;
  const int rowBase = blockIdx.x * 64 + wave * 16;
  load_w(wl, Wh1, (H2 * H) / 4);
  stage_bf16(al[wave], hT, nullptr, rowBase, M, H2);
  __syncthreads();
  v8f acc[8];
#pragma unroll
  for (int t = 0; t < 8; ++t) acc[t] = (v8f)(0.0f);
  wmma_accum(wl, al[wave], acc, 4, H2);

  const int lane = threadIdx.x & 31;
  const int half = lane >> 4, col = lane & 15;
  float p[8];
#pragma unroll
  for (int r = 0; r < 8; ++r) p[r] = 0.f;
#pragma unroll
  for (int t = 0; t < 8; ++t) {
    int n = t * 16 + col;
    float wv = w2[n], bv = b1[n];
#pragma unroll
    for (int r = 0; r < 8; ++r) {
      float u = fmaxf(acc[t][r] + bv, 0.f);
      p[r] += u * wv;
    }
  }
#pragma unroll
  for (int r = 0; r < 8; ++r)
    for (int off = 1; off < 16; off <<= 1)
      p[r] += __shfl_xor(p[r], off, 32);
  if (col == 0) {
    float bb = b2[0];
#pragma unroll
    for (int r = 0; r < 8; ++r) {
      int m = rowBase + half * 8 + r;
      if (m < M) out[m] = p[r] + bb;
    }
  }
}

// ---- prep / aggregation utility kernels ----

__global__ void zero_f32_kernel(float* __restrict__ p, long n) {
  long i = (long)blockIdx.x * blockDim.x + threadIdx.x;
  if (i < n) p[i] = 0.f;
}

__global__ void count_kernel(const int* __restrict__ e, float* __restrict__ cnt, int n) {
  int i = blockIdx.x * blockDim.x + threadIdx.x;
  if (i < n) atomicAdd(&cnt[e[i]], 1.0f);
}

// agg[e[i]][:] += h_tx[i][:]   (4 columns per thread, b64 load, f32 atomics)
__global__ void scatter_kernel(const unsigned* __restrict__ h,
                               const int* __restrict__ e,
                               float* __restrict__ agg, long total /* E*32 */) {
  long gid = (long)blockIdx.x * blockDim.x + threadIdx.x;
  if (gid >= total) return;
  long i = gid >> 5;
  int q = (int)(gid & 31);                  // uint-pair index within row
  uint2 d = ((const uint2*)(h + (size_t)i * H2))[q];
  float* dst = agg + (size_t)e[i] * H + q * 4;
  atomicAdd(dst + 0, bf16_f32((unsigned short)(d.x & 0xFFFFu)));
  atomicAdd(dst + 1, bf16_f32((unsigned short)(d.x >> 16)));
  atomicAdd(dst + 2, bf16_f32((unsigned short)(d.y & 0xFFFFu)));
  atomicAdd(dst + 3, bf16_f32((unsigned short)(d.y >> 16)));
}

// Pack f32 W[K][N=128] (optionally W+Wadd) into *fragment-major* bf16 pairs:
// out[((c*8 + t)*32 + lane)*8 + v] = {W[2k2][n], W[2k2+1][n]},
// k2 = c*16 + (lane>>4)*8 + v, n = t*16 + (lane&15).
__global__ void pack_w_kernel(const float* __restrict__ W, const float* __restrict__ Wadd,
                              unsigned* __restrict__ out, int K, int N) {
  int idx = blockIdx.x * blockDim.x + threadIdx.x;
  int K2 = K >> 1;
  if (idx >= K2 * N) return;
  int v    = idx & 7;
  int lane = (idx >> 3) & 31;
  int t    = (idx >> 8) & 7;
  int c    = idx >> 11;
  int k2 = c * 16 + (lane >> 4) * 8 + v;
  int n  = t * 16 + (lane & 15);
  float lo = W[(size_t)(2 * k2) * N + n];
  float hi = W[(size_t)(2 * k2 + 1) * N + n];
  if (Wadd) { lo += Wadd[(size_t)(2 * k2) * N + n]; hi += Wadd[(size_t)(2 * k2 + 1) * N + n]; }
  out[idx] = pk_bf16(lo, hi);
}

__global__ void add_bias_kernel(const float* __restrict__ a, const float* __restrict__ b,
                                float* __restrict__ out, int n) {
  int i = blockIdx.x * blockDim.x + threadIdx.x;
  if (i < n) out[i] = a[i] + b[i];
}

// ---------------------------------------------------------------------------
// Host launcher
// ---------------------------------------------------------------------------
extern "C" void kernel_launch(void* const* d_in, const int* in_sizes, int n_in,
                              void* d_out, int out_size, void* d_ws, size_t ws_size,
                              hipStream_t stream) {
  (void)in_sizes; (void)n_in; (void)out_size; (void)ws_size;
  const float* tx_x         = (const float*)d_in[0];
  const int*   e_card       = (const int*)d_in[3];
  const int*   e_merch      = (const int*)d_in[5];
  const float* card_emb     = (const float*)d_in[7];
  const float* merch_emb    = (const float*)d_in[8];
  const float* card_proj_W  = (const float*)d_in[9];
  const float* card_proj_b  = (const float*)d_in[10];
  const float* merch_proj_W = (const float*)d_in[11];
  const float* merch_proj_b = (const float*)d_in[12];
  const float* tx_W         = (const float*)d_in[13];
  const float* tx_b         = (const float*)d_in[14];
  const float* conv_Wl      = (const float*)d_in[15];
  const float* conv_bl      = (const float*)d_in[16];
  const float* conv_Wr      = (const float*)d_in[17];
  const float* h1_W         = (const float*)d_in[18];
  const float* h1_b         = (const float*)d_in[19];
  const float* h2_W         = (const float*)d_in[20];
  const float* h2_b         = (const float*)d_in[21];
  float* logits             = (float*)d_out;

  char* ws = (char*)d_ws;
  size_t off = 0;
  auto alloc = [&](size_t bytes) -> void* {
    void* p = ws + off;
    off = (off + bytes + 255) & ~(size_t)255;
    return p;
  };

  unsigned short* hT[2] = {(unsigned short*)alloc((size_t)NTX * H * 2),
                           (unsigned short*)alloc((size_t)NTX * H * 2)};
  unsigned short* hC[2] = {(unsigned short*)alloc((size_t)NCARD * H * 2),
                           (unsigned short*)alloc((size_t)NCARD * H * 2)};
  unsigned short* hM[2] = {(unsigned short*)alloc((size_t)NMERCH * H * 2),
                           (unsigned short*)alloc((size_t)NMERCH * H * 2)};
  float* aggC = (float*)alloc((size_t)NCARD * H * 4);
  float* aggM = (float*)alloc((size_t)NMERCH * H * 4);
  float* cntC = (float*)alloc((size_t)NCARD * 4);
  float* cntM = (float*)alloc((size_t)NMERCH * 4);

  const size_t WB128 = (size_t)H2 * H * 4;       // packed 128x128
  const size_t WB64  = (size_t)32 * H * 4;       // packed 64x128
  unsigned* pEncTx = (unsigned*)alloc(WB128);
  unsigned* pEncC  = (unsigned*)alloc(WB64);
  unsigned* pEncM  = (unsigned*)alloc(WB64);
  unsigned* pWl0[2], *pWl2[2], *pWrc[2], *pWl1[2], *pWr1[2], *pWl3[2], *pWr3[2];
  float* btx[2];
  for (int l = 0; l < 2; ++l) {
    pWl0[l] = (unsigned*)alloc(WB128);
    pWl2[l] = (unsigned*)alloc(WB128);
    pWrc[l] = (unsigned*)alloc(WB128);
    pWl1[l] = (unsigned*)alloc(WB128);
    pWr1[l] = (unsigned*)alloc(WB128);
    pWl3[l] = (unsigned*)alloc(WB128);
    pWr3[l] = (unsigned*)alloc(WB128);
    btx[l]  = (float*)alloc(H * 4);
  }
  unsigned* pWh1 = (unsigned*)alloc(WB128);

  auto packW = [&](const float* W, const float* Wadd, unsigned* dst, int K) {
    int n = (K / 2) * H;
    pack_w_kernel<<<(n + 255) / 256, 256, 0, stream>>>(W, Wadd, dst, K, H);
  };
  const size_t HH = (size_t)H * H;

  // ---- weight prep ----
  packW(tx_W, nullptr, pEncTx, H);
  packW(card_proj_W, nullptr, pEncC, 64);
  packW(merch_proj_W, nullptr, pEncM, 64);
  for (int l = 0; l < 2; ++l) {
    const float* Wl = conv_Wl + (size_t)l * 4 * HH;
    const float* Wr = conv_Wr + (size_t)l * 4 * HH;
    const float* bl = conv_bl + (size_t)l * 4 * H;
    packW(Wl + 0 * HH, nullptr, pWl0[l], H);
    packW(Wl + 2 * HH, nullptr, pWl2[l], H);
    packW(Wr + 0 * HH, Wr + 2 * HH, pWrc[l], H);   // Wr0 + Wr2 combined
    packW(Wl + 1 * HH, nullptr, pWl1[l], H);
    packW(Wr + 1 * HH, nullptr, pWr1[l], H);
    packW(Wl + 3 * HH, nullptr, pWl3[l], H);
    packW(Wr + 3 * HH, nullptr, pWr3[l], H);
    add_bias_kernel<<<1, H, 0, stream>>>(bl + 0 * H, bl + 2 * H, btx[l], H);
  }
  packW(h1_W, nullptr, pWh1, H);

  // ---- degree counts (constant across layers) ----
  zero_f32_kernel<<<(NCARD + 255) / 256, 256, 0, stream>>>(cntC, NCARD);
  zero_f32_kernel<<<(NMERCH + 255) / 256, 256, 0, stream>>>(cntM, NMERCH);
  count_kernel<<<(NEDGE + 255) / 256, 256, 0, stream>>>(e_card, cntC, NEDGE);
  count_kernel<<<(NEDGE + 255) / 256, 256, 0, stream>>>(e_merch, cntM, NEDGE);

  // ---- input encoders ----
  const int gTx = (NTX + 63) / 64, gC = (NCARD + 63) / 64, gM = (NMERCH + 63) / 64;
  encoder_kernel<<<gTx, 128, 0, stream>>>(tx_x, pEncTx, tx_b, hT[0], NTX, 64, 1);
  encoder_kernel<<<gC, 128, 0, stream>>>(card_emb, pEncC, card_proj_b, hC[0], NCARD, 32, 0);
  encoder_kernel<<<gM, 128, 0, stream>>>(merch_emb, pEncM, merch_proj_b, hM[0], NMERCH, 32, 0);

  // ---- 2 SAGE layers ----
  int cur = 0;
  const long totE = (long)NEDGE * 32;            // 4 cols per thread
  const int gScat = (int)((totE + 255) / 256);
  for (int l = 0; l < 2; ++l) {
    int nxt = cur ^ 1;
    zero_f32_kernel<<<(int)(((long)NCARD * H + 255) / 256), 256, 0, stream>>>(aggC, (long)NCARD * H);
    zero_f32_kernel<<<(int)(((long)NMERCH * H + 255) / 256), 256, 0, stream>>>(aggM, (long)NMERCH * H);
    scatter_kernel<<<gScat, 256, 0, stream>>>((const unsigned*)hT[cur], e_card, aggC, totE);
    scatter_kernel<<<gScat, 256, 0, stream>>>((const unsigned*)hT[cur], e_merch, aggM, totE);

    tx_layer_kernel<<<gTx, 128, 0, stream>>>(
        (const unsigned*)hC[cur], (const unsigned*)hM[cur], (const unsigned*)hT[cur],
        e_card, e_merch, pWl0[l], pWl2[l], pWrc[l], btx[l], hT[nxt], NTX);
    dst_layer_kernel<<<gC, 128, 0, stream>>>(
        aggC, cntC, (const unsigned*)hC[cur], pWl1[l], pWr1[l],
        conv_bl + ((size_t)l * 4 + 1) * H, hC[nxt], NCARD);
    dst_layer_kernel<<<gM, 128, 0, stream>>>(
        aggM, cntM, (const unsigned*)hM[cur], pWl3[l], pWr3[l],
        conv_bl + ((size_t)l * 4 + 3) * H, hM[nxt], NMERCH);
    cur = nxt;
  }

  // ---- classification head ----
  head_kernel<<<gTx, 128, 0, stream>>>((const unsigned*)hT[cur], pWh1, h1_b,
                                       h2_W, h2_b, logits, NTX);
}
